// LSTM_AdjMatrixDistribution_74869869904308
// MI455X (gfx1250) — matrix operand order repaired and checked by reading, not compiled, not run on hardware
//
#include <hip/hip_runtime.h>

typedef __attribute__((ext_vector_type(16))) _Float16 v16h;
typedef __attribute__((ext_vector_type(8)))  _Float16 v8h;
typedef __attribute__((ext_vector_type(8)))  float    v8f;

#define HIDDEN 48
#define GATES  192            // 4*HIDDEN
#define KCAT   96             // [x | h] concat, = 3 * 32 (WMMA K)
#define LAYERS 3
#define BATCH  1024
#define TSTEPS 992            // N_NODES*(N_NODES-1)
#define WAVES_PER_BLOCK 4
#define BLOCKS 16             // 16 blocks * 4 waves * 16 rows = 1024 = BATCH

// ---- dynamic LDS layout (bytes) ----
#define OFF_WCAT 0
#define SZ_WCAT  (LAYERS*GATES*KCAT*2)               // 110592 (f16 fused [w_ih|w_hh])
#define OFF_BIAS (OFF_WCAT + SZ_WCAT)
#define SZ_BIAS  (LAYERS*GATES*4)                    // 2304 (b_ih + b_hh)
#define OFF_EW   (OFF_BIAS + SZ_BIAS)
#define OFF_EB   (OFF_EW + HIDDEN*4)
#define OFF_PW   (OFF_EB + HIDDEN*4)
#define OFF_SBUF (OFF_PW + HIDDEN*4)
#define SZ_SBUF  (WAVES_PER_BLOCK*16*4)              // per-wave sample feedback
#define OFF_XH   (OFF_SBUF + SZ_SBUF)                // 32B aligned
#define SZ_XH    (WAVES_PER_BLOCK*LAYERS*16*KCAT*2)  // per-wave [x|h] A-tiles
#define SMEM_BYTES (OFF_XH + SZ_XH)                  // ~147.5 KB < 320 KB/WGP

// Combine two contiguous 8-half LDS slices into one WMMA 16-half fragment.
__device__ __forceinline__ v16h load_frag(const _Float16* p0, const _Float16* p1) {
  v8h lo = *(const v8h*)p0;
  v8h hi = *(const v8h*)p1;
  v16h r;
#pragma unroll
  for (int i = 0; i < 8; ++i) { r[i] = lo[i]; r[i + 8] = hi[i]; }
  return r;
}

__device__ __forceinline__ float sigmoid_f(float x) {
  return __builtin_amdgcn_rcpf(1.f + __expf(-x));
}
__device__ __forceinline__ float tanh_f(float x) {
  float t = __expf(2.f * x);                  // tanh(x) = 1 - 2/(e^{2x}+1)
  return 1.f - 2.f * __builtin_amdgcn_rcpf(t + 1.f);
}

__global__ __launch_bounds__(128, 1)
void lstm_adj_kernel(const float* __restrict__ embed_w, const float* __restrict__ embed_b,
                     const float* __restrict__ w_ih,    const float* __restrict__ w_hh,
                     const float* __restrict__ b_ih,    const float* __restrict__ b_hh,
                     const float* __restrict__ proj_w,  const float* __restrict__ proj_b,
                     const float* __restrict__ h0,      const float* __restrict__ c0,
                     const float* __restrict__ init_input,
                     const float* __restrict__ gumbel,
                     float* __restrict__ out)
{
  extern __shared__ char smem[];
  _Float16* Wcat   = (_Float16*)(smem + OFF_WCAT);
  float*    biasS  = (float*)(smem + OFF_BIAS);
  float*    ewS    = (float*)(smem + OFF_EW);
  float*    ebS    = (float*)(smem + OFF_EB);
  float*    pwS    = (float*)(smem + OFF_PW);
  float*    sbufA  = (float*)(smem + OFF_SBUF);
  _Float16* xhAll  = (_Float16*)(smem + OFF_XH);

  const int tid = threadIdx.x;

  // ---- one-time cooperative preload: f32 weights -> f16 LDS, fused [w_ih | w_hh] ----
  for (int i = tid; i < LAYERS*GATES*KCAT; i += blockDim.x) {
    int l   = i / (GATES*KCAT);
    int rem = i - l*(GATES*KCAT);
    int n   = rem / KCAT;
    int k   = rem - n*KCAT;
    float v = (k < HIDDEN) ? w_ih[(l*GATES + n)*HIDDEN + k]
                           : w_hh[(l*GATES + n)*HIDDEN + (k - HIDDEN)];
    Wcat[i] = (_Float16)v;
  }
  for (int i = tid; i < LAYERS*GATES; i += blockDim.x) biasS[i] = b_ih[i] + b_hh[i];
  for (int i = tid; i < HIDDEN; i += blockDim.x) {
    ewS[i] = embed_w[i]; ebS[i] = embed_b[i]; pwS[i] = proj_w[i];
  }
  __syncthreads();   // waves are fully independent after this point

  const int wave  = tid >> 5;
  const int lane  = tid & 31;
  const int laneM = lane & 15;          // A: row M / B,C: col N
  const int hiH   = lane >> 4;          // 0: lanes 0-15, 1: lanes 16-31
  const int batchBase = (blockIdx.x * WAVES_PER_BLOCK + wave) * 16;

  _Float16* xh   = xhAll + wave * (LAYERS*16*KCAT); // 3 A-tiles of 16x96 f16
  float*    sbuf = sbufA + wave * 16;

  // ---- init state: c in VGPRs (C-layout local), h in LDS f16 ----
  float cs[LAYERS][3][8];
#pragma unroll
  for (int l = 0; l < LAYERS; ++l)
#pragma unroll
    for (int t3 = 0; t3 < 3; ++t3) {
      float c0v = c0[l*HIDDEN + 16*t3 + laneM];     // broadcast across batch rows
      _Float16 h0h = (_Float16)h0[l*HIDDEN + 16*t3 + laneM];
#pragma unroll
      for (int r = 0; r < 8; ++r) {
        cs[l][t3][r] = c0v;
        xh[l*(16*KCAT) + (r + hiH*8)*KCAT + HIDDEN + 16*t3 + laneM] = h0h;
      }
    }
  if (lane < 16) {
    sbuf[lane] = init_input[0];
    float* ob = out + (size_t)(batchBase + lane) * (32*32);
    for (int n = 0; n < 32; ++n) ob[n*33] = 0.f;    // diagonal of adj = 0
  }
  const float pb = proj_b[0];

  // ---- 992 sequential timesteps, per-wave independent ----
  for (int t = 0; t < TSTEPS; ++t) {
    // embed: layer0 input half of A-tile = s*ew + eb
    for (int i = lane; i < 16*HIDDEN; i += 32) {
      int b = i / HIDDEN;
      int k = i - b*HIDDEN;
      xh[b*KCAT + k] = (_Float16)(sbuf[b] * ewS[k] + ebS[k]);
    }

#pragma unroll
    for (int l = 0; l < LAYERS; ++l) {
      const _Float16* xl = xh   + l*(16*KCAT);
      const _Float16* wl = Wcat + l*(GATES*KCAT);
      const float*    bl = biasS + l*GATES;
      // next layer's input slot; last layer writes h3 into layer0 slot (scratch)
      _Float16* dst = (l < LAYERS-1) ? (xh + (l+1)*(16*KCAT)) : xh;

      // accumulators seeded with bias: all 8 rows of a tile share one N -> one scalar
      v8f acc[12];
#pragma unroll
      for (int n = 0; n < 12; ++n) {
        float bv = bl[16*n + laneM];
        v8f ci;
#pragma unroll
        for (int r = 0; r < 8; ++r) ci[r] = bv;
        acc[n] = ci;
      }
      // gates(16x192) = A(16x96) @ Wcat^T : 3 K-chunks x 12 N-tiles of WMMA
#pragma unroll
      for (int kc = 0; kc < 3; ++kc) {
        const _Float16* arow = xl + laneM*KCAT + kc*32 + hiH*8;   // A frag: 2x16B
        v16h afrag = load_frag(arow, arow + 16);
#pragma unroll
        for (int n = 0; n < 12; ++n) {
          const _Float16* brow = wl + (16*n + laneM)*KCAT + kc*32 + hiH*8;  // B frag: 2x16B
          v16h bfrag = load_frag(brow, brow + 16);
          acc[n] = __builtin_amdgcn_wmma_f32_16x16x32_f16(
              false, afrag, false, bfrag, (short)0, acc[n], false, false);
        }
      }
      // LSTM pointwise: i/f/g/o for one element sit in same lane+VGPR across tiles
#pragma unroll
      for (int t3 = 0; t3 < 3; ++t3) {
#pragma unroll
        for (int r = 0; r < 8; ++r) {
          float iv = sigmoid_f(acc[t3    ][r]);
          float fv = sigmoid_f(acc[t3 + 3][r]);
          float gv = tanh_f   (acc[t3 + 6][r]);
          float ov = sigmoid_f(acc[t3 + 9][r]);
          float cn = fv * cs[l][t3][r] + iv * gv;
          cs[l][t3][r] = cn;
          _Float16 hh = (_Float16)(ov * tanh_f(cn));
          int row = r + hiH*8;
          xh[l*(16*KCAT) + row*KCAT + HIDDEN + 16*t3 + laneM] = hh; // h-state for t+1
          dst[row*KCAT + 16*t3 + laneM] = hh;                       // next layer input
        }
      }
    }

    // projection + gumbel hard sample; argmax(log p + g0, log(1-p)+g1)
    // reduces exactly to sign(z + g0 - g1), ties -> class 0 -> s = 1
    if (lane < 16) {
      const _Float16* h3 = xh + lane*KCAT;   // layer0 input slot = h3 scratch
      float z = pb;
#pragma unroll
      for (int k = 0; k < HIDDEN; ++k) z += (float)h3[k] * pwS[k];
      const float* gp = gumbel + ((size_t)t * BATCH + batchBase + lane) * 2;
      float s = (z + gp[0] - gp[1] >= 0.f) ? 1.f : 0.f;
      int rw = t / 31;
      int ix = t - rw*31;
      int cl = ix + (ix >= rw ? 1 : 0);      // off-diagonal row-major enumeration
      out[(size_t)(batchBase + lane) * (32*32) + rw*32 + cl] = s;
      sbuf[lane] = s;                        // autoregressive feedback
    }
  }
}

extern "C" void kernel_launch(void* const* d_in, const int* in_sizes, int n_in,
                              void* d_out, int out_size, void* d_ws, size_t ws_size,
                              hipStream_t stream) {
  (void)in_sizes; (void)n_in; (void)out_size; (void)d_ws; (void)ws_size;
  const float* embed_w   = (const float*)d_in[0];
  const float* embed_b   = (const float*)d_in[1];
  const float* w_ih      = (const float*)d_in[2];
  const float* w_hh      = (const float*)d_in[3];
  const float* b_ih      = (const float*)d_in[4];
  const float* b_hh      = (const float*)d_in[5];
  const float* proj_w    = (const float*)d_in[6];
  const float* proj_b    = (const float*)d_in[7];
  const float* h0        = (const float*)d_in[8];
  const float* c0        = (const float*)d_in[9];
  const float* init_inp  = (const float*)d_in[10];
  const float* gumbel    = (const float*)d_in[11];
  float* out = (float*)d_out;

  hipFuncSetAttribute((const void*)lstm_adj_kernel,
                      hipFuncAttributeMaxDynamicSharedMemorySize, SMEM_BYTES);
  lstm_adj_kernel<<<dim3(BLOCKS), dim3(128), SMEM_BYTES, stream>>>(
      embed_w, embed_b, w_ih, w_hh, b_ih, b_hh, proj_w, proj_b,
      h0, c0, init_inp, gumbel, out);
}
